// PointEGNNModuleMSG_88923002896881
// MI455X (gfx1250) — compile-verified, matrix operand order checked by reading, not run on hardware
//
#include <hip/hip_runtime.h>
#include <hip/hip_bf16.h>

typedef __attribute__((ext_vector_type(16))) _Float16 v16h;
typedef __attribute__((ext_vector_type(8)))  float    v8f;

#define NB   8192   // points
#define MB   2048   // FPS centers
#define CB   64     // feature channels
#define HB   128    // hidden
#define BB   2      // batch

__device__ __forceinline__ float silu_f(float x) {
    return x / (1.0f + __expf(-x));
}

// -------------------------------------------------------------------------
// Pack weight matrix (K x 128, row-major f32) into WMMA-B f16 fragments:
// dst[((kt*8 + nt)*32 + lane)*16 + half], K zero-padded to KT*32.
// Fragment element (n in tile, kappa in tile): lane = (n&15) + 16*kappa[3],
// half = kappa[2:0] + 8*kappa[4].  Two scales packed back-to-back.
// -------------------------------------------------------------------------
__global__ void pack_b_frags(const float* __restrict__ src, _Float16* __restrict__ dst,
                             int Kact, int KT) {
    int per_scale = KT * 8 * 32 * 16;
    int t = blockIdx.x * blockDim.x + threadIdx.x;
    if (t >= 2 * per_scale) return;
    int i = t / per_scale;
    int r = t - i * per_scale;
    int half = r & 15;
    int lane = (r >> 4) & 31;
    int nt   = (r >> 9) & 7;
    int kt   = r >> 12;
    int n = nt * 16 + (lane & 15);
    int k = kt * 32 + (half >> 3) * 16 + ((lane >> 4) & 1) * 8 + (half & 7);
    float v = (k < Kact) ? src[(i * Kact + k) * HB + n] : 0.0f;
    dst[t] = (_Float16)v;
}

// -------------------------------------------------------------------------
// FPS: one 1024-thread block per batch. Points live in registers (8/thread),
// 3-float LDS broadcast of the last pick, shfl+LDS argmax with first-index
// tiebreak. Writes idx (ws + out), gathers dsamp_xyz to out.
// -------------------------------------------------------------------------
__global__ __launch_bounds__(1024)
void fps_kernel(const float* __restrict__ xyz, int* __restrict__ idxWs,
                int* __restrict__ idxOut, float* __restrict__ dsampOut) {
    int b   = blockIdx.x;
    int tid = threadIdx.x;
    const float* P = xyz + (size_t)b * NB * 3;

    float px[8], py[8], pz[8], mind[8];
#pragma unroll
    for (int s = 0; s < 8; ++s) {
        int p = tid + s * 1024;
        px[s] = P[p * 3 + 0];
        py[s] = P[p * 3 + 1];
        pz[s] = P[p * 3 + 2];
        mind[s] = 1e10f;
    }

    __shared__ float bcx, bcy, bcz;
    __shared__ float redV[32];
    __shared__ int   redI[32];
    __shared__ int   winIdx;
    __shared__ int   selIdx[MB];

    if (tid == 0) { bcx = P[0]; bcy = P[1]; bcz = P[2]; selIdx[0] = 0; }
    __syncthreads();

    for (int m = 1; m < MB; ++m) {
        float lx = bcx, ly = bcy, lz = bcz;
        float bestv = -1.0f; int besti = 0;
#pragma unroll
        for (int s = 0; s < 8; ++s) {
            float dx = px[s] - lx, dy = py[s] - ly, dz = pz[s] - lz;
            float d = dx * dx + dy * dy + dz * dz;
            mind[s] = fminf(mind[s], d);
            int pi = tid + s * 1024;
            if (mind[s] > bestv || (mind[s] == bestv && pi < besti)) {
                bestv = mind[s]; besti = pi;
            }
        }
        // wave32 reduce
#pragma unroll
        for (int off = 16; off >= 1; off >>= 1) {
            float ov = __shfl_down(bestv, off);
            int   oi = __shfl_down(besti, off);
            if (ov > bestv || (ov == bestv && oi < besti)) { bestv = ov; besti = oi; }
        }
        int wv = tid >> 5;
        if ((tid & 31) == 0) { redV[wv] = bestv; redI[wv] = besti; }
        __syncthreads();
        if (wv == 0) {
            bestv = redV[tid & 31]; besti = redI[tid & 31];
#pragma unroll
            for (int off = 16; off >= 1; off >>= 1) {
                float ov = __shfl_down(bestv, off);
                int   oi = __shfl_down(besti, off);
                if (ov > bestv || (ov == bestv && oi < besti)) { bestv = ov; besti = oi; }
            }
            if (tid == 0) { winIdx = besti; selIdx[m] = besti; }
        }
        __syncthreads();
        int wi = winIdx;
        if ((wi & 1023) == tid) {
            int s = wi >> 10;
            bcx = px[s]; bcy = py[s]; bcz = pz[s];
        }
        __syncthreads();
    }

    for (int m = tid; m < MB; m += 1024) {
        int p = selIdx[m];
        idxWs[b * MB + m]  = p;
        idxOut[b * MB + m] = p;
        dsampOut[(size_t)(b * MB + m) * 3 + 0] = P[p * 3 + 0];
        dsampOut[(size_t)(b * MB + m) * 3 + 1] = P[p * 3 + 1];
        dsampOut[(size_t)(b * MB + m) * 3 + 2] = P[p * 3 + 2];
    }
}

// -------------------------------------------------------------------------
// KNN top-32 (sorted ascending; top-16 is a prefix). Thread-per-center,
// points staged through LDS in 1024-point tiles; unrolled 32-deep
// compare-swap insertion guarded by the current worst key.
// -------------------------------------------------------------------------
__global__ __launch_bounds__(128)
void knn_kernel(const float* __restrict__ xyz, const int* __restrict__ idxWs,
                int* __restrict__ nidx) {
    int b = blockIdx.y;
    int m = blockIdx.x * 128 + threadIdx.x;
    int tid = threadIdx.x;
    const float* P = xyz + (size_t)b * NB * 3;

    int ci = idxWs[b * MB + m];
    float cx = P[ci * 3 + 0], cy = P[ci * 3 + 1], cz = P[ci * 3 + 2];

    float key[32]; int kid[32];
#pragma unroll
    for (int j = 0; j < 32; ++j) { key[j] = 3.0e38f; kid[j] = 0; }

    __shared__ float tile[1024 * 3];
    for (int t0 = 0; t0 < NB; t0 += 1024) {
        __syncthreads();
        for (int e = tid; e < 1024 * 3; e += 128) tile[e] = P[t0 * 3 + e];
        if (t0 + 1024 < NB)
            __builtin_prefetch(P + (t0 + 1024) * 3 + tid * 24, 0, 0);
        __syncthreads();
        for (int p = 0; p < 1024; ++p) {
            float dx = tile[p * 3 + 0] - cx;
            float dy = tile[p * 3 + 1] - cy;
            float dz = tile[p * 3 + 2] - cz;
            float d = dx * dx + dy * dy + dz * dz;
            if (d < key[31]) {
                float dd = d; int ii = t0 + p;
#pragma unroll
                for (int j = 0; j < 32; ++j) {
                    if (dd < key[j]) {
                        float tv = key[j]; key[j] = dd; dd = tv;
                        int   ti = kid[j]; kid[j] = ii; ii = ti;
                    }
                }
            }
        }
    }
#pragma unroll
    for (int j = 0; j < 32; ++j)
        nidx[((size_t)(b * MB + m) << 5) + j] = kid[j];
}

// -------------------------------------------------------------------------
// Edge EGNN kernel: 64 edge rows per 128-thread block (wave w = row-tile w).
// e_in (64 x 160, f16) built directly in WMMA-A fragment layout in LDS;
// GEMM1 (160->128) -> SiLU -> A-frags in LDS; GEMM2 (128->128) -> SiLU ->
// m_ij f32 in LDS (overlaying e_in); then coef / agg / x_shift.
// -------------------------------------------------------------------------
__global__ __launch_bounds__(128)
void edge_kernel(const float* __restrict__ xyz, const float* __restrict__ feat,
                 const int* __restrict__ idxWs, const int* __restrict__ nidxWs,
                 const _Float16* __restrict__ we1p, const float* __restrict__ be1,
                 const _Float16* __restrict__ we2p, const float* __restrict__ be2,
                 const float* __restrict__ wx, const float* __restrict__ bxp,
                 const float* __restrict__ dsamp, float* __restrict__ xshift,
                 float* __restrict__ aggS, int scale, int kLog2) {
    const int k   = 1 << kLog2;
    const int cpb = 64 >> kLog2;             // centers per block
    int tid  = threadIdx.x;
    int lane = tid & 31;
    int wv   = tid >> 5;

    int rowBase = blockIdx.x * 64;
    int b     = rowBase >> (11 + kLog2);     // rows per batch = M*k = 2^(11+kLog2)
    int mBase = (rowBase >> kLog2) & (MB - 1);

    __shared__ __align__(32) unsigned char smemRaw[32768 + 16384];
    _Float16* einA = (_Float16*)smemRaw;                  // 4*5*32*16 halfs (20KB)
    float*    mij  = (float*)smemRaw;                     // 64*128 f32 (32KB) overlay
    _Float16* m1A  = (_Float16*)(smemRaw + 32768);        // 4*4*32*16 halfs (16KB)
    __shared__ float relb[64 * 3];
    __shared__ float coefb[64];
    __shared__ float d2sh[64];
    __shared__ int   nidsh[64];
    __shared__ int   cidsh[64];

    // per-row neighbor geometry
    if (tid < 64) {
        int r  = tid;
        int cc = r >> kLog2;
        int j  = r & (k - 1);
        int m  = mBase + cc;
        int nid = nidxWs[((size_t)(b * MB + m) << 5) + j];
        nidsh[r] = nid;
        cidsh[r] = idxWs[b * MB + m];
        float cx = dsamp[(size_t)(b * MB + m) * 3 + 0];
        float cy = dsamp[(size_t)(b * MB + m) * 3 + 1];
        float cz = dsamp[(size_t)(b * MB + m) * 3 + 2];
        float rx = cx - xyz[((size_t)b * NB + nid) * 3 + 0];
        float ry = cy - xyz[((size_t)b * NB + nid) * 3 + 1];
        float rz = cz - xyz[((size_t)b * NB + nid) * 3 + 2];
        relb[r * 3 + 0] = rx; relb[r * 3 + 1] = ry; relb[r * 3 + 2] = rz;
        d2sh[r] = rx * rx + ry * ry + rz * rz;
    }
    __syncthreads();

    // build e_in = [h_i | h_j | d2 | 0-pad] as A fragments (64 rows x 160 K)
    for (int e = tid; e < 64 * 160; e += 128) {
        int r = e / 160;
        int c = e - r * 160;
        float v;
        if (c < 64)       v = feat[((size_t)(b * CB + c)) * NB + cidsh[r]];
        else if (c < 128) v = feat[((size_t)(b * CB + (c - 64))) * NB + nidsh[r]];
        else if (c == 128) v = d2sh[r];
        else               v = 0.0f;
        int rt = r >> 4, rr = r & 15;
        int kt = c >> 5, k5 = c & 31;
        int ln = rr + ((k5 >> 3) & 1) * 16;
        int hf = (k5 & 7) + (k5 >> 4) * 8;
        einA[(((rt * 5 + kt) * 32) + ln) * 16 + hf] = (_Float16)v;
    }
    __syncthreads();

    // GEMM1: (64 x 160) x (160 x 128), SiLU, store to A frags for GEMM2
    const v16h* Bw1 = (const v16h*)(we1p + (size_t)scale * (5 * 8 * 32 * 16));
    int hiRow = (lane >> 4) * 8;
#pragma unroll
    for (int nt = 0; nt < 8; ++nt) {
        v8f acc = {};
#pragma unroll
        for (int kt = 0; kt < 5; ++kt) {
            v16h a = *(const v16h*)&einA[(((wv * 5) + kt) * 32 + lane) * 16];
            v16h bf = Bw1[(kt * 8 + nt) * 32 + lane];
            acc = __builtin_amdgcn_wmma_f32_16x16x32_f16(false, a, false, bf,
                                                         (short)0, acc, false, false);
        }
        int col0 = nt * 16 + (lane & 15);
        float bias = be1[scale * HB + col0];
#pragma unroll
        for (int g = 0; g < 8; ++g) {
            float x = acc[g] + bias;
            float s = silu_f(x);
            int rowL = g + hiRow;
            int kt2 = col0 >> 5, k5 = col0 & 31;
            int ln2 = rowL + ((k5 >> 3) & 1) * 16;
            int hf2 = (k5 & 7) + (k5 >> 4) * 8;
            m1A[(((wv * 4) + kt2) * 32 + ln2) * 16 + hf2] = (_Float16)s;
        }
    }
    __syncthreads();   // einA dead from here; mij may overlay it

    // GEMM2: (64 x 128) x (128 x 128), SiLU, store m_ij f32 to LDS
    const v16h* Bw2 = (const v16h*)(we2p + (size_t)scale * (4 * 8 * 32 * 16));
#pragma unroll
    for (int nt = 0; nt < 8; ++nt) {
        v8f acc = {};
#pragma unroll
        for (int kt = 0; kt < 4; ++kt) {
            v16h a = *(const v16h*)&m1A[(((wv * 4) + kt) * 32 + lane) * 16];
            v16h bf = Bw2[(kt * 8 + nt) * 32 + lane];
            acc = __builtin_amdgcn_wmma_f32_16x16x32_f16(false, a, false, bf,
                                                         (short)0, acc, false, false);
        }
        int col0 = nt * 16 + (lane & 15);
        float bias = be2[scale * HB + col0];
#pragma unroll
        for (int g = 0; g < 8; ++g) {
            float x = acc[g] + bias;
            mij[(wv * 16 + g + hiRow) * HB + col0] = silu_f(x);
        }
    }
    __syncthreads();

    // coef_r = m_ij[r] . wx + bx
    if (tid < 64) {
        const float* wxp = wx + scale * HB;
        float s = 0.0f;
#pragma unroll 8
        for (int h = 0; h < HB; ++h) s += mij[tid * HB + h] * wxp[h];
        coefb[tid] = s + bxp[scale];
    }
    __syncthreads();

    // agg[c][h] = sum_j m_ij
    for (int e = tid; e < cpb * HB; e += 128) {
        int cc = e >> 7, h = e & 127;
        float s = 0.0f;
        for (int j = 0; j < k; ++j) s += mij[((cc << kLog2) + j) * HB + h];
        int m = mBase + cc;
        aggS[((size_t)(b * MB + m)) * HB + h] = s;
    }
    // x_shift = dsamp + mean_j(rel * coef)
    if (tid < cpb * 3) {
        int cc = tid / 3, d = tid - cc * 3;
        float s = 0.0f;
        for (int j = 0; j < k; ++j) {
            int r = (cc << kLog2) + j;
            s += relb[r * 3 + d] * coefb[r];
        }
        int m = mBase + cc;
        xshift[((size_t)(b * 2 * MB) + scale * MB + m) * 3 + d] =
            dsamp[(size_t)(b * MB + m) * 3 + d] + s / (float)k;
    }
}

// -------------------------------------------------------------------------
// Node MLP: rows = centers (64/block). [h_i | agg] (192) ->128 SiLU ->128.
// Output stored transposed into feats (B, S*O, M).
// -------------------------------------------------------------------------
__global__ __launch_bounds__(128)
void node_kernel(const float* __restrict__ feat, const int* __restrict__ idxWs,
                 const float* __restrict__ aggS,
                 const _Float16* __restrict__ wh1p, const float* __restrict__ bh1,
                 const _Float16* __restrict__ wh2p, const float* __restrict__ bh2,
                 float* __restrict__ outFeats, int scale) {
    int tid  = threadIdx.x;
    int lane = tid & 31;
    int wv   = tid >> 5;
    int rowBase = blockIdx.x * 64;
    int b     = rowBase >> 11;
    int mBase = rowBase & (MB - 1);

    __shared__ __align__(32) _Float16 hinA[4 * 6 * 32 * 16];  // 24KB
    __shared__ __align__(32) _Float16 m1A[4 * 4 * 32 * 16];   // 16KB

    for (int e = tid; e < 64 * 192; e += 128) {
        int r = e / 192;
        int c = e - r * 192;
        int m = mBase + r;
        float v;
        if (c < 64) v = feat[((size_t)(b * CB + c)) * NB + idxWs[b * MB + m]];
        else        v = aggS[((size_t)(b * MB + m)) * HB + (c - 64)];
        int rt = r >> 4, rr = r & 15;
        int kt = c >> 5, k5 = c & 31;
        int ln = rr + ((k5 >> 3) & 1) * 16;
        int hf = (k5 & 7) + (k5 >> 4) * 8;
        hinA[(((rt * 6 + kt) * 32) + ln) * 16 + hf] = (_Float16)v;
    }
    __syncthreads();

    const v16h* Bw1 = (const v16h*)(wh1p + (size_t)scale * (6 * 8 * 32 * 16));
    int hiRow = (lane >> 4) * 8;
#pragma unroll
    for (int nt = 0; nt < 8; ++nt) {
        v8f acc = {};
#pragma unroll
        for (int kt = 0; kt < 6; ++kt) {
            v16h a = *(const v16h*)&hinA[(((wv * 6) + kt) * 32 + lane) * 16];
            v16h bf = Bw1[(kt * 8 + nt) * 32 + lane];
            acc = __builtin_amdgcn_wmma_f32_16x16x32_f16(false, a, false, bf,
                                                         (short)0, acc, false, false);
        }
        int col0 = nt * 16 + (lane & 15);
        float bias = bh1[scale * HB + col0];
#pragma unroll
        for (int g = 0; g < 8; ++g) {
            float s = silu_f(acc[g] + bias);
            int rowL = g + hiRow;
            int kt2 = col0 >> 5, k5 = col0 & 31;
            int ln2 = rowL + ((k5 >> 3) & 1) * 16;
            int hf2 = (k5 & 7) + (k5 >> 4) * 8;
            m1A[(((wv * 4) + kt2) * 32 + ln2) * 16 + hf2] = (_Float16)s;
        }
    }
    __syncthreads();

    const v16h* Bw2 = (const v16h*)(wh2p + (size_t)scale * (4 * 8 * 32 * 16));
#pragma unroll
    for (int nt = 0; nt < 8; ++nt) {
        v8f acc = {};
#pragma unroll
        for (int kt = 0; kt < 4; ++kt) {
            v16h a = *(const v16h*)&m1A[(((wv * 4) + kt) * 32 + lane) * 16];
            v16h bf = Bw2[(kt * 8 + nt) * 32 + lane];
            acc = __builtin_amdgcn_wmma_f32_16x16x32_f16(false, a, false, bf,
                                                         (short)0, acc, false, false);
        }
        int col0 = nt * 16 + (lane & 15);
        float bias = bh2[scale * HB + col0];
#pragma unroll
        for (int g = 0; g < 8; ++g) {
            int m = mBase + wv * 16 + g + hiRow;
            // feats (B, S*O, M)
            outFeats[((size_t)(b * 2 * HB) + scale * HB + col0) * MB + m] = acc[g] + bias;
        }
    }
}

// -------------------------------------------------------------------------
extern "C" void kernel_launch(void* const* d_in, const int* in_sizes, int n_in,
                              void* d_out, int out_size, void* d_ws, size_t ws_size,
                              hipStream_t stream) {
    const float* xyz  = (const float*)d_in[0];
    const float* feat = (const float*)d_in[1];
    const float* we1  = (const float*)d_in[2];
    const float* be1  = (const float*)d_in[3];
    const float* we2  = (const float*)d_in[4];
    const float* be2  = (const float*)d_in[5];
    const float* wx   = (const float*)d_in[6];
    const float* bx   = (const float*)d_in[7];
    const float* wh1  = (const float*)d_in[8];
    const float* bh1  = (const float*)d_in[9];
    const float* wh2  = (const float*)d_in[10];
    const float* bh2  = (const float*)d_in[11];

    float* out        = (float*)d_out;
    float* out_dsamp  = out;                               // (B, M, 3)
    float* out_shift  = out + (size_t)BB * MB * 3;         // (B, S*M, 3)
    float* out_feats  = out_shift + (size_t)BB * 2 * MB * 3; // (B, S*O, M)
    int*   out_idx    = (int*)(out_feats + (size_t)BB * 2 * HB * MB); // (B, M)

    unsigned char* ws = (unsigned char*)d_ws;
    size_t o = 0;
    auto carve = [&](size_t bytes) {
        void* p = ws + o;
        o = (o + bytes + 255) & ~(size_t)255;
        return p;
    };
    int*      ws_idx  = (int*)carve((size_t)BB * MB * 4);
    int*      ws_nidx = (int*)carve((size_t)BB * MB * 32 * 4);
    _Float16* we1p    = (_Float16*)carve((size_t)2 * 5 * 8 * 32 * 16 * 2);
    _Float16* we2p    = (_Float16*)carve((size_t)2 * 4 * 8 * 32 * 16 * 2);
    _Float16* wh1p    = (_Float16*)carve((size_t)2 * 6 * 8 * 32 * 16 * 2);
    _Float16* wh2p    = (_Float16*)carve((size_t)2 * 4 * 8 * 32 * 16 * 2);
    float*    ws_agg  = (float*)carve((size_t)2 * BB * MB * HB * 4);   // per-scale agg

    // 1. pre-pack weights into WMMA-B f16 fragment layout
    pack_b_frags<<<(2 * 5 * 8 * 32 * 16 + 255) / 256, 256, 0, stream>>>(we1, we1p, 129, 5);
    pack_b_frags<<<(2 * 4 * 8 * 32 * 16 + 255) / 256, 256, 0, stream>>>(we2, we2p, 128, 4);
    pack_b_frags<<<(2 * 6 * 8 * 32 * 16 + 255) / 256, 256, 0, stream>>>(wh1, wh1p, 192, 6);
    pack_b_frags<<<(2 * 4 * 8 * 32 * 16 + 255) / 256, 256, 0, stream>>>(wh2, wh2p, 128, 4);

    // 2. FPS (writes idx + dsamp_xyz)
    fps_kernel<<<BB, 1024, 0, stream>>>(xyz, ws_idx, out_idx, out_dsamp);

    // 3. KNN top-32 per center
    knn_kernel<<<dim3(MB / 128, BB), 128, 0, stream>>>(xyz, ws_idx, ws_nidx);

    // 4. Edge EGNN per scale (WMMA MLPs), writes agg + x_shift
    float* agg0 = ws_agg;
    float* agg1 = ws_agg + (size_t)BB * MB * HB;
    edge_kernel<<<(BB * MB * 16) / 64, 128, 0, stream>>>(
        xyz, feat, ws_idx, ws_nidx, we1p, be1, we2p, be2, wx, bx,
        out_dsamp, out_shift, agg0, 0, 4);
    edge_kernel<<<(BB * MB * 32) / 64, 128, 0, stream>>>(
        xyz, feat, ws_idx, ws_nidx, we1p, be1, we2p, be2, wx, bx,
        out_dsamp, out_shift, agg1, 1, 5);

    // 5. Node MLP per scale (WMMA), writes feats
    node_kernel<<<(BB * MB) / 64, 128, 0, stream>>>(
        feat, ws_idx, agg0, wh1p, bh1, wh2p, bh2, out_feats, 0);
    node_kernel<<<(BB * MB) / 64, 128, 0, stream>>>(
        feat, ws_idx, agg1, wh1p, bh1, wh2p, bh2, out_feats, 1);
}
